// ScaledDotProductWithBoxExtrAtt_71433896067280
// MI455X (gfx1250) — matrix operand order repaired and checked by reading, not compiled, not run on hardware
//
#include <hip/hip_runtime.h>
#include <hip/hip_bf16.h>

// Problem constants (fixed by reference)
#define B_ 16
#define N_ 512
#define D_ 512
#define H_ 8
#define DK_ 64
#define DV_ 64
#define BN_ (B_ * N_)      // 8192 rows in projection GEMMs
#define HD_ (H_ * DK_)     // 512 cols

typedef __attribute__((ext_vector_type(16))) _Float16 v16h;
typedef __attribute__((ext_vector_type(8)))  float    v8f;
typedef __attribute__((ext_vector_type(4)))  float    v4f;
typedef __attribute__((ext_vector_type(4)))  _Float16 v4h;

__device__ inline v8f wmma_f16(v16h a, v16h b, v8f c) {
  // v_wmma_f32_16x16x32_f16: D = A(16x32) x B(32x16) + C
  return __builtin_amdgcn_wmma_f32_16x16x32_f16(
      /*neg_a=*/false, a, /*neg_b=*/false, b,
      /*c_mod=*/(short)0, c, /*reuse_a=*/false, /*reuse_b=*/false);
}

// A operand (16x32, 16-bit, row-major source).
// ISA layout: lane L -> row m = L&15; VGPR v: k = (v/4)*16 + (L>=16 ? 8:0) + (v%4)*2 (+pair)
__device__ inline v16h load_a16(const _Float16* p, int ld, int row0, int k0) {
  const int L = threadIdx.x & 31;
  const int m = L & 15;
  const int hi = L >> 4;
  v16h a;
#pragma unroll
  for (int v = 0; v < 8; ++v) {
    const int k = k0 + (v >> 2) * 16 + hi * 8 + (v & 3) * 2;
    const _Float16* q = p + (size_t)(row0 + m) * ld + k;
    a[2 * v]     = q[0];
    a[2 * v + 1] = q[1];
  }
  return a;
}

// A operand from the in-LDS att2 probability buffer.
// Probabilities for row m, col c live at half-index m*1024 + (c>>8)*512 + (c&255)
// (f16 written over the low half of each lane-half's 1KB f32 sub-region).
__device__ inline v16h load_a16_probs(const _Float16* base, int k0) {
  const int L = threadIdx.x & 31;
  const int m = L & 15;
  const int hi = L >> 4;
  v16h a;
#pragma unroll
  for (int v = 0; v < 8; ++v) {
    const int k = k0 + (v >> 2) * 16 + hi * 8 + (v & 3) * 2;
    const int idx = m * 1024 + ((k >> 8) * 512) + (k & 255);
    a[2 * v]     = base[idx];
    a[2 * v + 1] = base[idx + 1];
  }
  return a;
}

// B operand (32x16, 16-bit), source is B^T stored row-major: bt[n][k].
// ISA layout: lane L -> col n = L&15; VGPR v: k = (v/4)*16 + (L>=16 ? 8:0) + (v%4)*2 (+pair)
__device__ inline v16h load_bt16(const _Float16* bt, int ld, int col0, int k0) {
  const int L = threadIdx.x & 31;
  const int n = L & 15;
  const int hi = L >> 4;
  v16h b;
#pragma unroll
  for (int v = 0; v < 8; ++v) {
    const int k = k0 + (v >> 2) * 16 + hi * 8 + (v & 3) * 2;
    const _Float16* q = bt + (size_t)(col0 + n) * ld + k;
    b[2 * v]     = q[0];
    b[2 * v + 1] = q[1];
  }
  return b;
}

// ---------------- elementwise converts ----------------

__global__ void cvt_f32_to_f16_v4(const float* __restrict__ in,
                                  _Float16* __restrict__ out, int n4) {
  int i = blockIdx.x * blockDim.x + threadIdx.x;
  if (i < n4) {
    v4f x = ((const v4f*)in)[i];
    ((v4h*)out)[i] = __builtin_convertvector(x, v4h);
  }
}

// out[c*R + r] = (f16) in[r*C + c]   (stores W^T for contiguous-K B operands)
__global__ void transpose_cvt(const float* __restrict__ in,
                              _Float16* __restrict__ out, int R, int C) {
  int i = blockIdx.x * blockDim.x + threadIdx.x;
  if (i >= R * C) return;
  int r = i / C, c = i % C;
  out[(size_t)c * R + r] = (_Float16)in[i];
}

// ---------------- WMMA GEMM: [8192,512] x [512,512] ----------------
// One wave computes a 16x64 output tile (4 accumulators share the A operand
// each K-step -> 4x less A traffic).
// mode 0: f16 row-major out (Q/K projections), scale applied after bias
// mode 1: f16 out, V-transposed store into [B,H,DV,N]
// mode 2: f32 row-major out, non-temporal (final projection -> d_out)
__global__ void gemm16(const _Float16* __restrict__ A,
                       const _Float16* __restrict__ BT,
                       const float* __restrict__ bias,
                       float scale, int mode,
                       _Float16* __restrict__ outH,
                       float* __restrict__ outF) {
  const int wave = blockIdx.x * (blockDim.x >> 5) + (threadIdx.x >> 5);
  const int mt = wave >> 3;       // 512 M-tiles
  const int ng = wave & 7;        // 8 groups of 4 N-tiles (64 cols)
  const int L = threadIdx.x & 31;
  const int lm = L & 15;
  const int hi = L >> 4;

  v8f c0 = {}, c1 = {}, c2 = {}, c3 = {};
#pragma unroll
  for (int s = 0; s < 16; ++s) {  // K = 512 = 16 * 32
    v16h a  = load_a16(A, D_, mt * 16, s * 32);
    v16h b0 = load_bt16(BT, D_, (ng * 4 + 0) * 16, s * 32);
    v16h b1 = load_bt16(BT, D_, (ng * 4 + 1) * 16, s * 32);
    v16h b2 = load_bt16(BT, D_, (ng * 4 + 2) * 16, s * 32);
    v16h b3 = load_bt16(BT, D_, (ng * 4 + 3) * 16, s * 32);
    c0 = wmma_f16(a, b0, c0);
    c1 = wmma_f16(a, b1, c1);
    c2 = wmma_f16(a, b2, c2);
    c3 = wmma_f16(a, b3, c3);
  }

  v8f acc[4] = {c0, c1, c2, c3};
#pragma unroll
  for (int t = 0; t < 4; ++t) {
    const int Nc = (ng * 4 + t) * 16 + lm;
    const float bv = bias[Nc];
#pragma unroll
    for (int j = 0; j < 8; ++j) {
      const int Mr = mt * 16 + j + hi * 8;
      const float val = (acc[t][j] + bv) * scale;
      if (mode == 0) {
        outH[(size_t)Mr * HD_ + Nc] = (_Float16)val;
      } else if (mode == 1) {
        const int b_ = Mr >> 9, tk = Mr & (N_ - 1);
        const int h_ = Nc >> 6, dv = Nc & (DV_ - 1);
        outH[((size_t)(b_ * H_ + h_) * DV_ + dv) * N_ + tk] = (_Float16)val;
      } else {
        __builtin_nontemporal_store(val, &outF[(size_t)Mr * D_ + Nc]);
      }
    }
  }
}

// ---------------- fused attention ----------------
// 1 wave per (b, h, 16-row q-tile), 32 KB LDS. Scores in LDS, double softmax
// fused with log(box) and ExtrAtt (each streamed once, NT: 268 MB > 192 MB L2),
// f16 att2 written in place over the score rows, fed back through WMMA.
__global__ void attn_kernel(const _Float16* __restrict__ qh,   // [B*N, H*DK] (pre-scaled)
                            const _Float16* __restrict__ kh,   // [B*N, H*DK]
                            const _Float16* __restrict__ vT,   // [B, H, DV, N]
                            const float* __restrict__ box,     // [B, H, N, N]
                            const float* __restrict__ extr,    // [B, H, N, N]
                            _Float16* __restrict__ attn_out) { // [B*N, H*DV]
  __shared__ float sc[16][N_];   // 32 KB (also holds f16 att2 in place)

  const int wg = blockIdx.x;            // 0..4095
  const int qt = wg & 31;
  const int h  = (wg >> 5) & 7;
  const int b  = wg >> 8;
  const int L  = threadIdx.x;
  const int lm = L & 15;
  const int hi = L >> 4;

  const _Float16* qbase = qh + (size_t)b * N_ * HD_ + h * DK_;
  const _Float16* kbase = kh + (size_t)b * N_ * HD_ + h * DK_;
  const size_t attbase = (((size_t)(b * H_ + h) * N_) + qt * 16) * N_;
  const float* boxb = box + attbase;
  const float* extb = extr + attbase;

  // A operand (q rows) loaded once: DK=64 -> two K=32 slabs
  v16h a0 = load_a16(qbase, HD_, qt * 16, 0);
  v16h a1 = load_a16(qbase, HD_, qt * 16, 32);

  // scores = q k^T + log(clip(box,1e-6))   (box: streamed once, NT)
  for (int kt = 0; kt < 32; ++kt) {
    v16h b0 = load_bt16(kbase, HD_, kt * 16, 0);
    v16h b1 = load_bt16(kbase, HD_, kt * 16, 32);
    v8f c = {};
    c = wmma_f16(a0, b0, c);
    c = wmma_f16(a1, b1, c);
#pragma unroll
    for (int j = 0; j < 8; ++j) {
      const int Mr = j + hi * 8;
      const int Nc = kt * 16 + lm;
      const float bxr = __builtin_nontemporal_load(&boxb[(size_t)Mr * N_ + Nc]);
      const float bx = fmaxf(bxr, 1e-6f);
      sc[Mr][Nc] = c[j] + __logf(bx);
    }
  }
  __syncthreads();

  // softmaxes over 512 keys; 2 lanes per row, float4-vectorized (64 iters each)
  {
    const int r = lm;
    v4f* rowv = (v4f*)&sc[r][0] + hi * 64;                      // this half's 64 v4f
    const v4f* extv = (const v4f*)&extb[(size_t)r * N_] + hi * 64;

    // softmax #1: max
    float mx = -3.4e38f;
    for (int j = 0; j < 64; ++j) {
      v4f x = rowv[j];
      mx = fmaxf(mx, fmaxf(fmaxf(x[0], x[1]), fmaxf(x[2], x[3])));
    }
    mx = fmaxf(mx, __shfl_xor(mx, 16));
    // softmax #1: exp + sum (in place)
    float sum = 0.f;
    for (int j = 0; j < 64; ++j) {
      v4f x = rowv[j], e;
      e[0] = __expf(x[0] - mx); e[1] = __expf(x[1] - mx);
      e[2] = __expf(x[2] - mx); e[3] = __expf(x[3] - mx);
      rowv[j] = e;
      sum += (e[0] + e[1]) + (e[2] + e[3]);
    }
    sum += __shfl_xor(sum, 16);
    const float inv = 1.f / sum;
    // logits #2 = w_mn*inv + ExtrAtt   (ExtrAtt: streamed once, NT, b128)
    for (int j = 0; j < 64; ++j) {
      v4f ex = __builtin_nontemporal_load(&extv[j]);
      rowv[j] = rowv[j] * inv + ex;
    }
    // softmax #2: max
    float mx2 = -3.4e38f;
    for (int j = 0; j < 64; ++j) {
      v4f x = rowv[j];
      mx2 = fmaxf(mx2, fmaxf(fmaxf(x[0], x[1]), fmaxf(x[2], x[3])));
    }
    mx2 = fmaxf(mx2, __shfl_xor(mx2, 16));
    // softmax #2: exp + sum (in place)
    float sum2 = 0.f;
    for (int j = 0; j < 64; ++j) {
      v4f x = rowv[j], e;
      e[0] = __expf(x[0] - mx2); e[1] = __expf(x[1] - mx2);
      e[2] = __expf(x[2] - mx2); e[3] = __expf(x[3] - mx2);
      rowv[j] = e;
      sum2 += (e[0] + e[1]) + (e[2] + e[3]);
    }
    sum2 += __shfl_xor(sum2, 16);
    const float inv2 = 1.f / sum2;
    // emit f16 probs in place: writer (8 B/iter at byte hi*1024 + 8j) trails
    // this lane's f32 reads (16 B/iter at byte hi*1024 + 16j); halves disjoint.
    v4h* p2v = (v4h*)((_Float16*)&sc[r][0] + hi * 512);
    for (int j = 0; j < 64; ++j) {
      v4f x = rowv[j] * inv2;
      p2v[j] = __builtin_convertvector(x, v4h);
    }
  }
  __syncthreads();

  // out = att2 @ v : [16,512] x [512,64], A from in-LDS f16 probs
  const _Float16* probs = (const _Float16*)&sc[0][0];
  const _Float16* vb = vT + (size_t)(b * H_ + h) * DV_ * N_; // [DV, N] for (b,h)
#pragma unroll
  for (int nt = 0; nt < 4; ++nt) {
    v8f c = {};
#pragma unroll
    for (int s = 0; s < 16; ++s) {
      v16h aa = load_a16_probs(probs, s * 32);
      v16h bb = load_bt16(vb, N_, nt * 16, s * 32);
      c = wmma_f16(aa, bb, c);
    }
#pragma unroll
    for (int j = 0; j < 8; ++j) {
      const int Mr = qt * 16 + j + hi * 8;
      attn_out[((size_t)(b * N_ + Mr)) * (H_ * DV_) + h * DV_ + nt * 16 + lm] =
          (_Float16)c[j];
    }
  }
}

extern "C" void kernel_launch(void* const* d_in, const int* in_sizes, int n_in,
                              void* d_out, int out_size, void* d_ws, size_t ws_size,
                              hipStream_t stream) {
  const float* queries = (const float*)d_in[0];
  const float* keys    = (const float*)d_in[1];
  const float* values  = (const float*)d_in[2];
  const float* box     = (const float*)d_in[3];
  const float* extr    = (const float*)d_in[4];
  const float* Wq = (const float*)d_in[5];  const float* bq = (const float*)d_in[6];
  const float* Wk = (const float*)d_in[7];  const float* bk = (const float*)d_in[8];
  const float* Wv = (const float*)d_in[9];  const float* bv = (const float*)d_in[10];
  const float* Wo = (const float*)d_in[11]; const float* bo = (const float*)d_in[12];
  float* out = (float*)d_out;

  char* ws = (char*)d_ws;
  const size_t SZ_IN = (size_t)BN_ * D_ * sizeof(_Float16);   // 8 MB
  const size_t SZ_W  = (size_t)D_ * D_ * sizeof(_Float16);    // 0.5 MB
  _Float16* qf   = (_Float16*)(ws);
  _Float16* kf   = (_Float16*)(ws + SZ_IN);
  _Float16* vf   = (_Float16*)(ws + 2 * SZ_IN);
  _Float16* WqT  = (_Float16*)(ws + 3 * SZ_IN);
  _Float16* WkT  = (_Float16*)(ws + 3 * SZ_IN + SZ_W);
  _Float16* WvT  = (_Float16*)(ws + 3 * SZ_IN + 2 * SZ_W);
  _Float16* WoT  = (_Float16*)(ws + 3 * SZ_IN + 3 * SZ_W);
  _Float16* qh   = (_Float16*)(ws + 3 * SZ_IN + 4 * SZ_W);
  _Float16* kh   = (_Float16*)(ws + 4 * SZ_IN + 4 * SZ_W);
  _Float16* vTq  = (_Float16*)(ws + 5 * SZ_IN + 4 * SZ_W);
  _Float16* aout = (_Float16*)(ws + 6 * SZ_IN + 4 * SZ_W);

  const int n4 = (BN_ * D_) / 4;  // 1,048,576 float4 groups
  cvt_f32_to_f16_v4<<<4096, 256, 0, stream>>>(queries, qf, n4);
  cvt_f32_to_f16_v4<<<4096, 256, 0, stream>>>(keys,    kf, n4);
  cvt_f32_to_f16_v4<<<4096, 256, 0, stream>>>(values,  vf, n4);
  transpose_cvt<<<1024, 256, 0, stream>>>(Wq, WqT, D_, D_);
  transpose_cvt<<<1024, 256, 0, stream>>>(Wk, WkT, D_, D_);
  transpose_cvt<<<1024, 256, 0, stream>>>(Wv, WvT, D_, D_);
  transpose_cvt<<<1024, 256, 0, stream>>>(Wo, WoT, D_, D_);

  // 4096 tile-waves (512 Mtiles x 8 Ngroups), 8 waves per 256-thread block
  const float qscale = 0.125f;  // 1/sqrt(DK)
  gemm16<<<512, 256, 0, stream>>>(qf, WqT, bq, qscale, 0, qh,  nullptr);
  gemm16<<<512, 256, 0, stream>>>(kf, WkT, bk, 1.0f,   0, kh,  nullptr);
  gemm16<<<512, 256, 0, stream>>>(vf, WvT, bv, 1.0f,   1, vTq, nullptr);

  attn_kernel<<<B_ * H_ * (N_ / 16), 32, 0, stream>>>(qh, kh, vTq, box, extr, aout);

  gemm16<<<512, 256, 0, stream>>>(aout, WoT, bo, 1.0f, 2, nullptr, out);
}
